// UniqueEq3NetMini_40355512713225
// MI455X (gfx1250) — matrix operand order repaired and checked by reading, not compiled, not run on hardware
//
#include <hip/hip_runtime.h>

// B=16, K=32, H=W=28, HID=OUT=64, NOPS=8.  Output: (B,1) fp32 = 16 floats.

typedef __attribute__((ext_vector_type(16))) _Float16 v16h;
typedef __attribute__((ext_vector_type(8)))  float    v8f;
typedef __attribute__((ext_vector_type(8)))  short    v8s;

union V16H { v16h v; _Float16 h[16]; };
union V8F  { v8f  v; float    f[8];  };
union V8S  { v8s  v; short    s[8];  };

// Compile-time im2col offset table: d (=ci*9+tap) -> LDS offset in padded image.
struct OffTab { short v[288]; };
static constexpr OffTab makeOffTab() {
  OffTab t{};
  for (int d = 0; d < 288; ++d) {
    const int ci = d / 9, tap = d % 9;
    t.v[d] = (short)(ci * 256 + (tap / 3) * 16 + (tap % 3));
  }
  return t;
}
__device__ __constant__ OffTab g_offTab = makeOffTab();

// ---------------- conv1 (1->32, 3x3 SAME) + relu + maxpool2 ----------------
__global__ __launch_bounds__(256) void eq3_conv1(const float* __restrict__ x,
    const float* __restrict__ w, const float* __restrict__ bias,
    float* __restrict__ h1) {
  const int n = blockIdx.x;                 // 512 images
  __shared__ float img[784];
  __shared__ float ws[288];
  __shared__ float bs[32];
  for (int t = threadIdx.x; t < 784; t += 256) img[t] = x[n * 784 + t];
  for (int t = threadIdx.x; t < 288; t += 256) ws[t] = w[t];
  if (threadIdx.x < 32) bs[threadIdx.x] = bias[threadIdx.x];
  __syncthreads();
  for (int t = threadIdx.x; t < 32 * 196; t += 256) {
    const int c = t / 196, r = t % 196, py = r / 14, px = r % 14;
    float m = 0.0f;                         // max over relu'd values (>=0)
    for (int sy = 0; sy < 2; ++sy)
      for (int sx = 0; sx < 2; ++sx) {
        const int y = 2 * py + sy, xx = 2 * px + sx;
        float acc = bs[c];
        #pragma unroll
        for (int dy = 0; dy < 3; ++dy) {
          const int iy = y + dy - 1;
          if ((unsigned)iy < 28u) {
            #pragma unroll
            for (int dx = 0; dx < 3; ++dx) {
              const int ix = xx + dx - 1;
              if ((unsigned)ix < 28u)
                acc = fmaf(ws[c * 9 + dy * 3 + dx], img[iy * 28 + ix], acc);
            }
          }
        }
        m = fmaxf(m, acc);
      }
    h1[n * 6272 + t] = m;
  }
}

// ------ pack weights into ISA WMMA B-fragment layout (f16, per-lane rows) ----
// w0frag: 8 frags  (nt=0..3, kc=0..1) of eq_w[:,:,0]   (K=64 d,  N=64 e)
// w2frag: 36 frags (nt=0..3, kc=0..8) of conv2_w        (K=288,   N=64 cout)
// frag elem m of lane L: K = kc*32 + (m<8 ? (L>>4)*8 + m : 16 + (L>>4)*8 + m-8), N = nt*16 + (L&15)
__global__ __launch_bounds__(256) void eq3_packw(const float* __restrict__ eqw,
    const float* __restrict__ w2, _Float16* __restrict__ w0frag,
    _Float16* __restrict__ w2frag) {
  const int t = blockIdx.x * 256 + threadIdx.x;   // 22528 total
  if (t < 4096) {
    const int m = t & 15, lane = (t >> 4) & 31, fid = t >> 9;  // fid = nt*2+kc
    const int kc = fid & 1, nt = fid >> 1;
    const int d = kc * 32 + (m < 8 ? (lane >> 4) * 8 + m : 16 + (lane >> 4) * 8 + (m - 8));
    const int e = nt * 16 + (lane & 15);
    w0frag[t] = (_Float16)eqw[(d * 64 + e) * 8];
  } else if (t < 22528) {
    const int t2 = t - 4096;
    const int m = t2 & 15, lane = (t2 >> 4) & 31, fid = t2 >> 9; // fid = nt*9+kc
    const int kc = fid % 9, nt = fid / 9;
    const int k = kc * 32 + (m < 8 ? (lane >> 4) * 8 + m : 16 + (lane >> 4) * 8 + (m - 8)); // ci*9 + tap
    const int cout = nt * 16 + (lane & 15);
    w2frag[t2] = (_Float16)w2[cout * 288 + k];
  }
}

// -------- conv2 (32->64, 3x3 SAME) via WMMA implicit GEMM + relu + pool ------
// Per image: M=196 pixels (13 tiles), N=64 cout (4 tiles), K=288 (9 chunks).
__global__ __launch_bounds__(256) void eq3_conv2w(const float* __restrict__ h1,
    const _Float16* __restrict__ w2frag, const float* __restrict__ bias,
    float* __restrict__ h2) {
  const int n = blockIdx.x;                 // 512 images
  __shared__ _Float16 imgp[32 * 256];       // 16 KB: 16x16 zero-padded per ci
  __shared__ _Float16 convs[208 * 64];      // 26 KB: [pixel][cout], 12 pad rows
  for (int t = threadIdx.x; t < 8192; t += 256) imgp[t] = (_Float16)0.f;
  __syncthreads();
  for (int t = threadIdx.x; t < 6272; t += 256) {
    const int ci = t / 196, r = t % 196, y = r / 14, x = r % 14;
    imgp[ci * 256 + (y + 1) * 16 + (x + 1)] = (_Float16)h1[n * 6272 + t];
  }
  __syncthreads();

  const int wave = threadIdx.x >> 5;
  const int lane = threadIdx.x & 31;
  const int nc = lane & 15, hi = lane >> 4;
  const v16h* __restrict__ wv = (const v16h*)w2frag;
  const short* __restrict__ tab = g_offTab.v;

  float bnt[4];
  #pragma unroll
  for (int nt = 0; nt < 4; ++nt) bnt[nt] = bias[nt * 16 + nc];

  for (int mt = wave; mt < 13; mt += 8) {
    const int p = mt * 16 + nc;
    const int pc = p < 196 ? p : 195;       // clamp tail rows (pad rows unread)
    const int y = pc / 14, x = pc % 14;
    const _Float16* abase = &imgp[y * 16 + x];   // + table offset per element

    V8F C[4];
    #pragma unroll
    for (int nt = 0; nt < 4; ++nt)
      #pragma unroll
      for (int r = 0; r < 8; ++r) C[nt].f[r] = 0.f;

    #pragma unroll 1                         // keep live set small: no spills
    for (int kc = 0; kc < 9; ++kc) {
      const int dbase = kc * 32 + hi * 8;    // 16B-aligned in table
      V8S tLo, tHi;
      tLo.v = *(const v8s*)&tab[dbase];      // 8 consecutive d's: one b128
      tHi.v = *(const v8s*)&tab[dbase + 16];
      V16H Af;
      #pragma unroll
      for (int m = 0; m < 8; ++m) Af.h[m] = abase[tLo.s[m]];
      #pragma unroll
      for (int m = 0; m < 8; ++m) Af.h[8 + m] = abase[tHi.s[m]];
      #pragma unroll
      for (int nt = 0; nt < 4; ++nt) {
        V16H Bf; Bf.v = wv[(nt * 9 + kc) * 32 + lane];
        C[nt].v = __builtin_amdgcn_wmma_f32_16x16x32_f16(
            false, Af.v, false, Bf.v, (short)0, C[nt].v, false, false);
      }
    }
    // Unconditional stores: rows 196..207 land in pad rows, never read.
    #pragma unroll
    for (int nt = 0; nt < 4; ++nt)
      #pragma unroll
      for (int r = 0; r < 8; ++r) {
        const int po = mt * 16 + hi * 8 + r; // C layout: M = r + 8*hi
        convs[po * 64 + nt * 16 + nc] = (_Float16)(C[nt].f[r] + bnt[nt]);
      }
  }
  __syncthreads();

  for (int t = threadIdx.x; t < 64 * 49; t += 256) {
    const int c = t / 49, r = t % 49, py = r / 7, px = r % 7;
    float m = 0.0f;                          // relu folded into max with 0
    #pragma unroll
    for (int sy = 0; sy < 2; ++sy)
      #pragma unroll
      for (int sx = 0; sx < 2; ++sx)
        m = fmaxf(m, (float)convs[((2 * py + sy) * 14 + 2 * px + sx) * 64 + c]);
    h2[n * 3136 + t] = m;
  }
}

// ---------------- fc (3136 -> 64) + relu ----------------
__global__ __launch_bounds__(64) void eq3_fc(const float* __restrict__ h2,
    const float* __restrict__ fw, const float* __restrict__ fb,
    float* __restrict__ feat) {
  const int n = blockIdx.x;                 // 512 rows
  __shared__ float row[3136];
  for (int t = threadIdx.x; t < 3136; t += 64) row[t] = h2[n * 3136 + t];
  __syncthreads();
  const int e = threadIdx.x;
  float acc = fb[e];
  const float* wr = &fw[e * 3136];
  for (int t = 0; t < 3136; ++t) acc = fmaf(row[t], wr[t], acc);
  feat[n * 64 + e] = fmaxf(acc, 0.0f);
}

// --------- prep1: s = mean_i a_i ; SA = s*a ; G4/G5/G6/G7 ; zero x8acc ------
__global__ __launch_bounds__(256) void eq3_prep1(const float* __restrict__ feat,
    const float* __restrict__ eqw, float* __restrict__ SA,
    float* __restrict__ G4, float* __restrict__ G5, float* __restrict__ G6,
    float* __restrict__ G7, float* __restrict__ x8acc) {
  const int b = blockIdx.x;                 // 16
  __shared__ float A[2048];
  __shared__ float s1[64], s2[64], s3[64];
  for (int t = threadIdx.x; t < 2048; t += 256) A[t] = feat[b * 2048 + t];
  __syncthreads();
  if (threadIdx.x < 64) {
    const int d = threadIdx.x;
    float acc = 0.f;
    for (int i = 0; i < 32; ++i) acc += A[i * 64 + d];
    acc *= (1.0f / 32.0f);
    s1[d] = acc; s2[d] = acc * acc; s3[d] = acc * acc * acc;
    x8acc[b * 64 + d] = 0.f;
  }
  __syncthreads();
  for (int t = threadIdx.x; t < 2048; t += 256)
    SA[b * 2048 + t] = s1[t & 63] * A[t];
  for (int t = threadIdx.x; t < 2048; t += 256) {
    const int p = t >> 6, e = t & 63;
    float a4 = 0.f, a5 = 0.f, a6 = 0.f;
    for (int d = 0; d < 64; ++d) {
      const float f = s2[d] * A[p * 64 + d];
      const float* wb = &eqw[(d * 64 + e) * 8];
      a4 = fmaf(f, wb[4], a4);
      a5 = fmaf(f, wb[5], a5);
      a6 = fmaf(f, wb[6], a6);
    }
    G4[b * 2048 + t] = a4; G5[b * 2048 + t] = a5; G6[b * 2048 + t] = a6;
  }
  if (threadIdx.x < 64) {
    const int e = threadIdx.x;
    float a7 = 0.f;
    for (int d = 0; d < 64; ++d) a7 = fmaf(s3[d], eqw[(d * 64 + e) * 8 + 7], a7);
    G7[b * 64 + e] = a7;
  }
}

// --------- prep2: pairwise tensors G1'(col add), G2'(=G2+G6), G3 ----------
__global__ __launch_bounds__(64) void eq3_prep2(const float* __restrict__ feat,
    const float* __restrict__ SA, const float* __restrict__ eqw,
    const float* __restrict__ eqb, const float* __restrict__ G4,
    const float* __restrict__ G5, const float* __restrict__ G6,
    const float* __restrict__ G7, float* __restrict__ G1c,
    float* __restrict__ G2p, float* __restrict__ G3) {
  const int idx = blockIdx.x;               // 16*32*32
  const int b = idx >> 10, p = (idx >> 5) & 31, q = idx & 31;
  __shared__ float prod[64];
  const int e = threadIdx.x;
  prod[e] = SA[(b * 32 + p) * 64 + e] * feat[(b * 32 + q) * 64 + e];
  __syncthreads();
  float a1 = 0.f, a2 = 0.f, a3 = 0.f;
  for (int d = 0; d < 64; ++d) {
    const float f = prod[d];
    const float* wb = &eqw[(d * 64 + e) * 8];
    a1 = fmaf(f, wb[1], a1);
    a2 = fmaf(f, wb[2], a2);
    a3 = fmaf(f, wb[3], a3);
  }
  const int o = idx * 64 + e;
  G1c[o] = a1 + G4[(b * 32 + q) * 64 + e] + G5[(b * 32 + p) * 64 + e]
              + G7[b * 64 + e] + eqb[e];   // all i-invariant terms + bias
  G2p[o] = a2 + G6[(b * 32 + p) * 64 + e]; // indexed [b,i=p,k=q,e]
  G3[o]  = a3;                             // indexed [b,i=p,j=q,e]
}

// --------- main: per-(b,j,k) 32x64x64 WMMA GEMM + fused epilogue ----------
__global__ __launch_bounds__(256) void eq3_main(const float* __restrict__ feat,
    const _Float16* __restrict__ w0frag, const float* __restrict__ G1c,
    const float* __restrict__ G2p, const float* __restrict__ G3,
    float* __restrict__ x8acc) {
  const int b    = blockIdx.x >> 7;                               // 2048 / 16
  const int pair = ((blockIdx.x & 127) << 3) + (threadIdx.x >> 5);
  const int j = pair >> 5, k = pair & 31;
  const int lane = threadIdx.x & 31;
  const int nc = lane & 15, hi = lane >> 4;

  __shared__ float As[2048];                // A[b] : 32 x 64 fp32
  for (int t = threadIdx.x; t < 2048; t += 256) As[t] = feat[b * 2048 + t];
  __syncthreads();

  // Pre-swizzled B fragments: two b128 loads each instead of 16 scalar loads.
  const v16h* __restrict__ wv = (const v16h*)w0frag;
  V16H Bf[4][2];
  #pragma unroll
  for (int nt = 0; nt < 4; ++nt)
    #pragma unroll
    for (int kc = 0; kc < 2; ++kc)
      Bf[nt][kc].v = wv[(nt * 2 + kc) * 32 + lane];

  V8F C[2][4];
  #pragma unroll
  for (int mt = 0; mt < 2; ++mt)
    #pragma unroll
    for (int nt = 0; nt < 4; ++nt)
      #pragma unroll
      for (int r = 0; r < 8; ++r) C[mt][nt].f[r] = 0.f;

  const float* aj = &As[j * 64];
  const float* ak = &As[k * 64];

  #pragma unroll
  for (int kc = 0; kc < 2; ++kc) {
    #pragma unroll
    for (int mt = 0; mt < 2; ++mt) {
      const float* ar = &As[(mt * 16 + nc) * 64];
      V16H Af;
      #pragma unroll
      for (int m = 0; m < 16; ++m) {
        const int d = kc * 32 + (m < 8 ? hi * 8 + m : 16 + hi * 8 + (m - 8));
        Af.h[m] = (_Float16)(ar[d] * aj[d] * ak[d]);
      }
      #pragma unroll
      for (int nt = 0; nt < 4; ++nt)
        C[mt][nt].v = __builtin_amdgcn_wmma_f32_16x16x32_f16(
            false, Af.v, false, Bf[nt][kc].v, (short)0, C[mt][nt].v, false, false);
    }
  }

  // Epilogue: x6 = T0 + G1'(col) + G2'(i,k) + G3(i,j); relu; reduce over i.
  float partial[4] = {0.f, 0.f, 0.f, 0.f};
  #pragma unroll
  for (int nt = 0; nt < 4; ++nt) {
    const int e = nt * 16 + nc;
    const float colAdd = G1c[((b * 32 + j) * 32 + k) * 64 + e];
    #pragma unroll
    for (int mt = 0; mt < 2; ++mt) {
      #pragma unroll
      for (int r = 0; r < 8; ++r) {
        const int i = mt * 16 + hi * 8 + r;
        const float v = C[mt][nt].f[r] + colAdd
                      + G2p[((b * 32 + i) * 32 + k) * 64 + e]
                      + G3 [((b * 32 + i) * 32 + j) * 64 + e];
        partial[nt] += fmaxf(v, 0.f);
      }
    }
  }
  #pragma unroll
  for (int nt = 0; nt < 4; ++nt)
    atomicAdd(&x8acc[b * 64 + nt * 16 + nc], partial[nt]);
}

// --------- final: x9 = relu(mean) ; out = x9 @ out_w.T + out_b ----------
__global__ __launch_bounds__(64) void eq3_final(const float* __restrict__ x8acc,
    const float* __restrict__ ow, const float* __restrict__ ob,
    float* __restrict__ out) {
  const int b = threadIdx.x;
  if (b < 16) {
    float acc = ob[0];
    for (int e = 0; e < 64; ++e) {
      const float x9 = fmaxf(x8acc[b * 64 + e] * (1.0f / 32768.0f), 0.f);
      acc = fmaf(x9, ow[e], acc);
    }
    out[b] = acc;
  }
}

extern "C" void kernel_launch(void* const* d_in, const int* in_sizes, int n_in,
                              void* d_out, int out_size, void* d_ws, size_t ws_size,
                              hipStream_t stream) {
  const float* x       = (const float*)d_in[0];
  const float* conv1_w = (const float*)d_in[1];
  const float* conv1_b = (const float*)d_in[2];
  const float* conv2_w = (const float*)d_in[3];
  const float* conv2_b = (const float*)d_in[4];
  const float* fc_w    = (const float*)d_in[5];
  const float* fc_b    = (const float*)d_in[6];
  const float* eq_w    = (const float*)d_in[7];
  const float* eq_b    = (const float*)d_in[8];
  const float* out_w   = (const float*)d_in[9];
  const float* out_b   = (const float*)d_in[10];

  float* wsf  = (float*)d_ws;                     // ~31 MB total
  float* h1   = wsf;                              // 512*32*14*14
  float* h2   = h1 + 3211264;                     // 512*64*7*7
  float* feat = h2 + 1605632;                     // 512*64
  float* SA   = feat + 32768;                     // 16*32*64
  float* G4   = SA + 32768;
  float* G5   = G4 + 32768;
  float* G6   = G5 + 32768;
  float* G7   = G6 + 32768;                       // 16*64
  float* G1c  = G7 + 1024;                        // 16*32*32*64
  float* G2p  = G1c + 1048576;
  float* G3   = G2p + 1048576;
  float* x8   = G3 + 1048576;                     // 16*64
  _Float16* w0frag = (_Float16*)(x8 + 1024);      // 4096 halfs, swizzled
  _Float16* w2frag = w0frag + 4096;               // 18432 halfs, swizzled

  eq3_conv1<<<512, 256, 0, stream>>>(x, conv1_w, conv1_b, h1);
  eq3_packw<<<88, 256, 0, stream>>>(eq_w, conv2_w, w0frag, w2frag);
  eq3_conv2w<<<512, 256, 0, stream>>>(h1, w2frag, conv2_b, h2);
  eq3_fc<<<512, 64, 0, stream>>>(h2, fc_w, fc_b, feat);
  eq3_prep1<<<16, 256, 0, stream>>>(feat, eq_w, SA, G4, G5, G6, G7, x8);
  eq3_prep2<<<16384, 64, 0, stream>>>(feat, SA, eq_w, eq_b, G4, G5, G6, G7,
                                      G1c, G2p, G3);
  eq3_main<<<2048, 256, 0, stream>>>(feat, w0frag, G1c, G2p, G3, x8);
  eq3_final<<<1, 64, 0, stream>>>(x8, out_w, out_b, (float*)d_out);

  (void)in_sizes; (void)n_in; (void)out_size; (void)ws_size;
}
